// CustomTriangleMultiplicationOutgoing_17197049053184
// MI455X (gfx1250) — compile-verified, matrix operand order checked by reading, not run on hardware
//
#include <hip/hip_runtime.h>
#include <math.h>

#define N 384
#define D 128
#define C 128
#define NN (N * N)

typedef __attribute__((ext_vector_type(16))) _Float16 v16h;
typedef __attribute__((ext_vector_type(8)))  _Float16 v8h;
typedef __attribute__((ext_vector_type(4)))  _Float16 v4h;
typedef __attribute__((ext_vector_type(8)))  float    v8f;
typedef __attribute__((ext_vector_type(4)))  float    v4f;
typedef __attribute__((ext_vector_type(4)))  unsigned v4u;
typedef __attribute__((ext_vector_type(8)))  unsigned v8u;

// Load a 16x32 (f16) WMMA A/B fragment. ISA layout (CDNA5 ISA):
// lane L<16 : row M=L, K = {0..7, 16..23}   -> two contiguous 16B chunks
// lane L>=16: row M=L-16, K = {8..15, 24..31}
__device__ __forceinline__ v16h load_frag(const _Float16* base, int stride, int kbase) {
  const int lane = threadIdx.x & 31;
  const _Float16* p = base + (size_t)(lane & 15) * stride + kbase + ((lane >> 4) << 3);
  v16h v;
  ((v4f*)&v)[0] = *(const v4f*)p;
  ((v4f*)&v)[1] = *(const v4f*)(p + 16);
  return v;
}

// ---- CDNA5 async global->LDS copy (ASYNCcnt) ------------------------------
__device__ __forceinline__ void async_copy_b128(unsigned lds_off, const void* gptr) {
  asm volatile("global_load_async_to_lds_b128 %0, %1, off"
               :: "v"(lds_off), "v"(gptr) : "memory");
}
__device__ __forceinline__ void wait_async0() {
  asm volatile("s_wait_asynccnt 0" ::: "memory");
}

// ---- CDNA5 Tensor Data Mover: 2D tile global->LDS (TENSORcnt) -------------
// D# group0 (CDNA5 ISA): count=1, lds_addr, global_addr, type=2.
__device__ __forceinline__ v4u tdm_g0(unsigned lds_off, const void* gaddr) {
  const size_t ga = (size_t)gaddr;
  v4u g;
  g.x = 1u;                                   // count=1 (valid user descriptor)
  g.y = lds_off;                              // LDS byte address
  g.z = (unsigned)ga;                         // global_addr[31:0]
  g.w = (unsigned)(ga >> 32) | 0x80000000u;   // global_addr[56:32] | type=2<<30
  return g;
}
// D# group1 (8.4): data_size=1 (2B elems), 2D tensor dim0/dim1, tile0/tile1, stride0.
__device__ __forceinline__ v8u tdm_g1(unsigned dim0, unsigned dim1,
                                      unsigned tile0, unsigned tile1,
                                      unsigned stride0) {
  v8u g;
  g[0] = 0x00010000u;                          // wg_mask=0, data_size=1, no pad/iter
  g[1] = (dim0 & 0xffffu) << 16;               // atomic_addr=0 | tensor_dim0[15:0]
  g[2] = (dim0 >> 16) | ((dim1 & 0xffffu) << 16);
  g[3] = (dim1 >> 16) | (tile0 << 16);         // tensor_dim1[31:16] | tile_dim0
  g[4] = tile1;                                // tile_dim1 | tile_dim2=0
  g[5] = stride0;                              // tensor_dim0_stride[31:0]
  g[6] = 0u;                                   // stride0[47:32]=0 | stride1 lo=0
  g[7] = 0u;
  return g;
}
__device__ __forceinline__ void tdm_load_2d(v4u g0, v8u g1) {
  asm volatile("tensor_load_to_lds %0, %1" :: "s"(g0), "s"(g1) : "memory");
}

__device__ __forceinline__ float sigmoidf_(float x) { return 1.0f / (1.0f + __expf(-x)); }

// ---------------------------------------------------------------------------
// Kernel A: convert 6 weight matrices fp32 [128][128] -> f16 transposed
// ---------------------------------------------------------------------------
__global__ void convert_weights(const float* Wa, const float* Wga, const float* Wb,
                                const float* Wgb, const float* Wgo, const float* Wo,
                                _Float16* WaT, _Float16* WgaT, _Float16* WbT,
                                _Float16* WgbT, _Float16* WgoT, _Float16* WoT) {
  const float* src; _Float16* dst;
  switch (blockIdx.x) {
    case 0: src = Wa;  dst = WaT;  break;
    case 1: src = Wga; dst = WgaT; break;
    case 2: src = Wb;  dst = WbT;  break;
    case 3: src = Wgb; dst = WgbT; break;
    case 4: src = Wgo; dst = WgoT; break;
    default: src = Wo; dst = WoT;  break;
  }
  for (int idx = threadIdx.x; idx < 128 * 128; idx += blockDim.x) {
    int i = idx >> 7, j = idx & 127;
    dst[j * 128 + i] = (_Float16)src[i * 128 + j];
  }
}

// ---------------------------------------------------------------------------
// Kernel B: fused LayerNorm(z) + 5 projections (WMMA) + sigmoid gates + mask.
// Weights staged in LDS with one TDM descriptor each.
// ---------------------------------------------------------------------------
__global__ void __launch_bounds__(128) ln_proj_kernel(
    const float* __restrict__ z, const float* __restrict__ mask,
    const float* __restrict__ ng, const float* __restrict__ nb,
    const _Float16* __restrict__ WaT,  const float* __restrict__ ba,
    const _Float16* __restrict__ WgaT, const float* __restrict__ bga,
    const _Float16* __restrict__ WbT,  const float* __restrict__ bb,
    const _Float16* __restrict__ WgbT, const float* __restrict__ bgb,
    const _Float16* __restrict__ WgoT, const float* __restrict__ bgo,
    _Float16* __restrict__ left_t, _Float16* __restrict__ right_t,
    _Float16* __restrict__ g_t)
{
  __shared__ _Float16 zt[4][16 * 128];   // per-wave normalized tile (f16)
  __shared__ _Float16 wt[128 * 128];     // staged weight (transposed, f16)

  const int wave = threadIdx.x >> 5, lane = threadIdx.x & 31;
  const int r = lane & 15, hi = lane >> 4;
  const int m0 = blockIdx.x * 64 + wave * 16;

  // ---- LayerNorm over D=128: lane pair (r, r+16) splits the row in halves
  const float* zr = z + (size_t)(m0 + r) * D + hi * 64;
  float s = 0.f, s2 = 0.f;
#pragma unroll
  for (int t = 0; t < 16; ++t) {
    v4f v = ((const v4f*)zr)[t];
    s  += v.x + v.y + v.z + v.w;
    s2 += v.x * v.x + v.y * v.y + v.z * v.z + v.w * v.w;
  }
  s  += __shfl_xor(s, 16);
  s2 += __shfl_xor(s2, 16);
  const float mean = s * (1.0f / D);
  const float rstd = rsqrtf(s2 * (1.0f / D) - mean * mean + 1e-5f);
#pragma unroll
  for (int t = 0; t < 16; ++t) {
    v4f v  = ((const v4f*)zr)[t];
    v4f gv = ((const v4f*)(ng + hi * 64))[t];
    v4f bv = ((const v4f*)(nb + hi * 64))[t];
    v4h h;
    h.x = (_Float16)((v.x - mean) * rstd * gv.x + bv.x);
    h.y = (_Float16)((v.y - mean) * rstd * gv.y + bv.y);
    h.z = (_Float16)((v.z - mean) * rstd * gv.z + bv.z);
    h.w = (_Float16)((v.w - mean) * rstd * gv.w + bv.w);
    *(v4h*)&zt[wave][r * 128 + hi * 64 + t * 4] = h;
  }

  __syncthreads();
  v16h a[4];
#pragma unroll
  for (int kt = 0; kt < 4; ++kt) a[kt] = load_frag(zt[wave], 128, kt * 32);

  // Stage one 128x128 f16 weight with a single TDM descriptor (wave 0 issues).
  auto stage = [&](const _Float16* W) {
    __syncthreads();                       // previous GEMM done reading wt
    if (wave == 0) {
      tdm_load_2d(tdm_g0((unsigned)(size_t)wt, W), tdm_g1(128, 128, 128, 128, 128));
      __builtin_amdgcn_s_wait_tensorcnt(0);
    }
    __syncthreads();
  };

  // Software-pipelined over nt: next tile's B-frags loaded before current WMMAs.
  auto do_gemm = [&](v8f* acc) {
    v16h b0 = load_frag(wt, 128, 0);
    v16h b1 = load_frag(wt, 128, 32);
    v16h b2 = load_frag(wt, 128, 64);
    v16h b3 = load_frag(wt, 128, 96);
#pragma unroll
    for (int nt = 0; nt < 8; ++nt) {
      v16h c0 = b0, c1 = b1, c2 = b2, c3 = b3;
      if (nt < 7) {
        const _Float16* wb = wt + (nt + 1) * 16 * 128;
        b0 = load_frag(wb, 128, 0);
        b1 = load_frag(wb, 128, 32);
        b2 = load_frag(wb, 128, 64);
        b3 = load_frag(wb, 128, 96);
      }
      v8f c = {};
      c = __builtin_amdgcn_wmma_f32_16x16x32_f16(false, a[0], false, c0, (short)0, c, false, false);
      c = __builtin_amdgcn_wmma_f32_16x16x32_f16(false, a[1], false, c1, (short)0, c, false, false);
      c = __builtin_amdgcn_wmma_f32_16x16x32_f16(false, a[2], false, c2, (short)0, c, false, false);
      c = __builtin_amdgcn_wmma_f32_16x16x32_f16(false, a[3], false, c3, (short)0, c, false, false);
      acc[nt] = c;
    }
  };

  auto emit_gated = [&](const v8f* accA, const v8f* accG, const float* biasA,
                        const float* biasG, _Float16* out) {
    float mk[8];
#pragma unroll
    for (int q = 0; q < 8; ++q) mk[q] = mask[m0 + hi * 8 + q];
#pragma unroll
    for (int nt = 0; nt < 8; ++nt) {
      const int cidx = nt * 16 + r;
      const float bA = biasA[cidx], bG = biasG[cidx];
      v8h h;
#pragma unroll
      for (int q = 0; q < 8; ++q)
        h[q] = (_Float16)((accA[nt][q] + bA) * sigmoidf_(accG[nt][q] + bG) * mk[q]);
      *(v8h*)(out + (size_t)cidx * NN + m0 + hi * 8) = h;
    }
  };

  v8f acc0[8], acc1[8];

  stage(WaT);  do_gemm(acc0);
  stage(WgaT); do_gemm(acc1);
  emit_gated(acc0, acc1, ba, bga, left_t);

  stage(WbT);  do_gemm(acc0);
  stage(WgbT); do_gemm(acc1);
  emit_gated(acc0, acc1, bb, bgb, right_t);

  stage(WgoT); do_gemm(acc0);
#pragma unroll
  for (int nt = 0; nt < 8; ++nt) {
    const int didx = nt * 16 + r;
    const float bG = bgo[didx];
    v8h h;
#pragma unroll
    for (int q = 0; q < 8; ++q)
      h[q] = (_Float16)sigmoidf_(acc0[nt][q] + bG);
    *(v8h*)(g_t + (size_t)didx * NN + m0 + hi * 8) = h;
  }
}

// ---------------------------------------------------------------------------
// Kernel C: einsum bikc,bjkc->bijc == per-channel Out_c = L_c @ R_c^T.
// 128x128 block tile, 8 waves x (64x32). K in slabs of 32, double-buffered
// LDS filled by TDM 2D tile loads (wave 0 issues; overlap with WMMA).
// ---------------------------------------------------------------------------
__global__ void __launch_bounds__(256) einsum_kernel(
    const _Float16* __restrict__ left_t, const _Float16* __restrict__ right_t,
    float* __restrict__ z_out)
{
  __shared__ _Float16 at[2][128 * 32];
  __shared__ _Float16 bt[2][128 * 32];

  const int c = blockIdx.y;
  const int i0 = (blockIdx.x / 3) * 128, j0 = (blockIdx.x % 3) * 128;
  const _Float16* Lc = left_t + (size_t)c * NN;
  const _Float16* Rc = right_t + (size_t)c * NN;

  const int wave = threadIdx.x >> 5, lane = threadIdx.x & 31;
  const int wi = wave & 1, wj = wave >> 1;     // 2x4 waves: 64 rows x 32 cols each

  // TDM: 32(k)x128(rows) tile of a 384x384 f16 tensor -> dense LDS tile
  auto stage_tdm = [&](int buf, int k0) {
    tdm_load_2d(tdm_g0((unsigned)(size_t)&at[buf][0], Lc + (size_t)i0 * N + k0),
                tdm_g1(N, N, 32, 128, N));
    tdm_load_2d(tdm_g0((unsigned)(size_t)&bt[buf][0], Rc + (size_t)j0 * N + k0),
                tdm_g1(N, N, 32, 128, N));
  };

  v8f acc[4][2] = {};

  if (wave == 0) {
    stage_tdm(0, 0);
    __builtin_amdgcn_s_wait_tensorcnt(0);
  }
  __syncthreads();

  for (int step = 0; step < N / 32; ++step) {
    const int buf = step & 1;
    if (wave == 0 && step + 1 < N / 32) stage_tdm(buf ^ 1, (step + 1) * 32);

    v16h bfrag[2];
#pragma unroll
    for (int n = 0; n < 2; ++n)
      bfrag[n] = load_frag(&bt[buf][(wj * 32 + n * 16) * 32], 32, 0);
#pragma unroll
    for (int m = 0; m < 4; ++m) {
      v16h afrag = load_frag(&at[buf][(wi * 64 + m * 16) * 32], 32, 0);
#pragma unroll
      for (int n = 0; n < 2; ++n)
        acc[m][n] = __builtin_amdgcn_wmma_f32_16x16x32_f16(
            false, afrag, false, bfrag[n], (short)0, acc[m][n], false, false);
    }
    if (wave == 0) __builtin_amdgcn_s_wait_tensorcnt(0);
    __syncthreads();
  }

  const int r = lane & 15, hi = lane >> 4;
#pragma unroll
  for (int n = 0; n < 2; ++n) {
    const int j = j0 + wj * 32 + n * 16 + r;
#pragma unroll
    for (int m = 0; m < 4; ++m) {
#pragma unroll
      for (int q = 0; q < 8; ++q) {
        const int i = i0 + wi * 64 + m * 16 + hi * 8 + q;
        z_out[((size_t)i * N + j) * C + c] = acc[m][n][q];
      }
    }
  }
}

// ---------------------------------------------------------------------------
// Kernel D: LayerNorm over C + (z_out @ Wo + bo) * g  -> d_out (f32).
// Weight staged with CDNA5 async global->LDS copies (ASYNCcnt path).
// ---------------------------------------------------------------------------
__global__ void __launch_bounds__(128) ln_out_kernel(
    const float* __restrict__ z_out, const float* __restrict__ nog,
    const float* __restrict__ nob, const _Float16* __restrict__ WoT,
    const float* __restrict__ bo, const _Float16* __restrict__ g_t,
    float* __restrict__ out)
{
  __shared__ _Float16 zt[4][16 * 128];
  __shared__ _Float16 wt[128 * 128];

#pragma unroll 4
  for (int o = threadIdx.x; o < 128 * 128 / 8; o += 128)
    async_copy_b128((unsigned)(size_t)&wt[o * 8], WoT + o * 8);

  const int wave = threadIdx.x >> 5, lane = threadIdx.x & 31;
  const int r = lane & 15, hi = lane >> 4;
  const int m0 = blockIdx.x * 64 + wave * 16;

  const float* zr = z_out + (size_t)(m0 + r) * C + hi * 64;
  float s = 0.f, s2 = 0.f;
#pragma unroll
  for (int t = 0; t < 16; ++t) {
    v4f v = ((const v4f*)zr)[t];
    s  += v.x + v.y + v.z + v.w;
    s2 += v.x * v.x + v.y * v.y + v.z * v.z + v.w * v.w;
  }
  s  += __shfl_xor(s, 16);
  s2 += __shfl_xor(s2, 16);
  const float mean = s * (1.0f / C);
  const float rstd = rsqrtf(s2 * (1.0f / C) - mean * mean + 1e-5f);
#pragma unroll
  for (int t = 0; t < 16; ++t) {
    v4f v  = ((const v4f*)zr)[t];
    v4f gv = ((const v4f*)(nog + hi * 64))[t];
    v4f bv = ((const v4f*)(nob + hi * 64))[t];
    v4h h;
    h.x = (_Float16)((v.x - mean) * rstd * gv.x + bv.x);
    h.y = (_Float16)((v.y - mean) * rstd * gv.y + bv.y);
    h.z = (_Float16)((v.z - mean) * rstd * gv.z + bv.z);
    h.w = (_Float16)((v.w - mean) * rstd * gv.w + bv.w);
    *(v4h*)&zt[wave][r * 128 + hi * 64 + t * 4] = h;
  }
  wait_async0();
  __syncthreads();

  v16h a[4];
#pragma unroll
  for (int kt = 0; kt < 4; ++kt) a[kt] = load_frag(zt[wave], 128, kt * 32);

#pragma unroll
  for (int nt = 0; nt < 8; ++nt) {
    const _Float16* wb = wt + nt * 16 * 128;
    v16h b0 = load_frag(wb, 128, 0);
    v16h b1 = load_frag(wb, 128, 32);
    v16h b2 = load_frag(wb, 128, 64);
    v16h b3 = load_frag(wb, 128, 96);
    v8f acc = {};
    acc = __builtin_amdgcn_wmma_f32_16x16x32_f16(false, a[0], false, b0, (short)0, acc, false, false);
    acc = __builtin_amdgcn_wmma_f32_16x16x32_f16(false, a[1], false, b1, (short)0, acc, false, false);
    acc = __builtin_amdgcn_wmma_f32_16x16x32_f16(false, a[2], false, b2, (short)0, acc, false, false);
    acc = __builtin_amdgcn_wmma_f32_16x16x32_f16(false, a[3], false, b3, (short)0, acc, false, false);

    const int didx = nt * 16 + r;
    const float bO = bo[didx];
    v8h gh = *(const v8h*)(g_t + (size_t)didx * NN + m0 + hi * 8);
#pragma unroll
    for (int q = 0; q < 8; ++q)
      out[(size_t)(m0 + hi * 8 + q) * D + didx] = (acc[q] + bO) * (float)gh[q];
  }
}

// ---------------------------------------------------------------------------
extern "C" void kernel_launch(void* const* d_in, const int* in_sizes, int n_in,
                              void* d_out, int out_size, void* d_ws, size_t ws_size,
                              hipStream_t stream) {
  const float* z    = (const float*)d_in[0];
  const float* mask = (const float*)d_in[1];
  const float* ng   = (const float*)d_in[2];
  const float* nb   = (const float*)d_in[3];
  const float* nog  = (const float*)d_in[4];
  const float* nob  = (const float*)d_in[5];
  const float* Wa   = (const float*)d_in[6];
  const float* ba   = (const float*)d_in[7];
  const float* Wb   = (const float*)d_in[8];
  const float* bb   = (const float*)d_in[9];
  const float* Wga  = (const float*)d_in[10];
  const float* bga  = (const float*)d_in[11];
  const float* Wgb  = (const float*)d_in[12];
  const float* bgb  = (const float*)d_in[13];
  const float* Wo   = (const float*)d_in[14];
  const float* bo   = (const float*)d_in[15];
  const float* Wgo  = (const float*)d_in[16];
  const float* bgo  = (const float*)d_in[17];

  char* p = (char*)d_ws;
  _Float16* WaT  = (_Float16*)p; p += 128 * 128 * 2;
  _Float16* WgaT = (_Float16*)p; p += 128 * 128 * 2;
  _Float16* WbT  = (_Float16*)p; p += 128 * 128 * 2;
  _Float16* WgbT = (_Float16*)p; p += 128 * 128 * 2;
  _Float16* WgoT = (_Float16*)p; p += 128 * 128 * 2;
  _Float16* WoT  = (_Float16*)p; p += 128 * 128 * 2;
  _Float16* left_t  = (_Float16*)p; p += (size_t)C * NN * 2;
  _Float16* right_t = (_Float16*)p; p += (size_t)C * NN * 2;
  _Float16* g_t     = (_Float16*)p; p += (size_t)D * NN * 2;
  float* z_out = (float*)p;

  convert_weights<<<6, 256, 0, stream>>>(Wa, Wga, Wb, Wgb, Wgo, Wo,
                                         WaT, WgaT, WbT, WgbT, WgoT, WoT);

  ln_proj_kernel<<<NN / 64, 128, 0, stream>>>(
      z, mask, ng, nb, WaT, ba, WgaT, bga, WbT, bb, WgbT, bgb, WgoT, bgo,
      left_t, right_t, g_t);

  einsum_kernel<<<dim3(9, 128), 256, 0, stream>>>(left_t, right_t, z_out);

  ln_out_kernel<<<NN / 64, 128, 0, stream>>>(z_out, nog, nob, WoT, bo, g_t,
                                             (float*)d_out);
}
